// DenoiserBlock_75505525064337
// MI455X (gfx1250) — matrix operand order repaired and checked by reading, not compiled, not run on hardware
//
#include <hip/hip_runtime.h>
#include <hip/hip_bf16.h>

// Problem constants (reference: B=2, L=2048, D=1024, H=16, F=4096)
#define BB 2
#define LL 2048
#define DD 1024
#define HH 16
#define HD 64
#define FF 4096
#define BLROWS (BB * LL)   // 4096

typedef __attribute__((ext_vector_type(16))) __bf16         v16bf;
typedef __attribute__((ext_vector_type(16))) unsigned short v16u;
typedef __attribute__((ext_vector_type(8)))  float          v8f;

// ---------- scalar helpers ----------
__device__ __forceinline__ unsigned short f2bf(float f) {
  unsigned int u = __float_as_uint(f);
  u += 0x7FFFu + ((u >> 16) & 1u);               // round-to-nearest-even
  return (unsigned short)(u >> 16);
}
__device__ __forceinline__ float gelu_f(float x) {
  const float c = 0.7978845608028654f;           // sqrt(2/pi)
  return 0.5f * x * (1.0f + tanhf(c * (x + 0.044715f * x * x * x)));
}
__device__ __forceinline__ v8f v8f_zero() {
  v8f z;
#pragma unroll
  for (int i = 0; i < 8; ++i) z[i] = 0.0f;
  return z;
}
__device__ __forceinline__ v8f wmma_bf16(const v16u a, const v16u b, const v8f c) {
  return __builtin_amdgcn_wmma_f32_16x16x32_bf16(
      false, __builtin_bit_cast(v16bf, a),
      false, __builtin_bit_cast(v16bf, b),
      (short)0, c, false, false);
}

// ---------- CDNA5 async global->LDS copy (ISA ch.10 / 15.18, ASYNCcnt) ----------
__device__ __forceinline__ void async_b128(const void* g, void* l) {
  asm volatile("global_load_async_to_lds_b128 %0, %1, off"
               :
               : "v"((unsigned)(unsigned long long)l),
                 "v"((unsigned long long)g)
               : "memory");
}
#define WAIT_ASYNCCNT(imm) asm volatile("s_wait_asynccnt " #imm ::: "memory")

// ---------- fp32 [K][N] -> bf16 [N][K] transpose-convert (LDS tiled) ----------
__global__ __launch_bounds__(256) void cvt_t_kernel(const float* __restrict__ w,
                                                    unsigned short* __restrict__ o,
                                                    int K, int N) {
  __shared__ float tile[32][33];
  const int k0 = blockIdx.y * 32;
  const int n0 = blockIdx.x * 32;
  const int tx = threadIdx.x & 31;
  const int ty = threadIdx.x >> 5;               // 0..7
#pragma unroll
  for (int i = 0; i < 4; ++i) {
    int k = ty + i * 8;
    tile[k][tx] = w[(size_t)(k0 + k) * N + n0 + tx];
  }
  __syncthreads();
#pragma unroll
  for (int i = 0; i < 4; ++i) {
    int n = ty + i * 8;
    o[(size_t)(n0 + n) * K + k0 + tx] = f2bf(tile[tx][n]);
  }
}

// ---------- tp = gelu(time_emb) @ Wt + bt  (B x 2D, tiny) ----------
__global__ __launch_bounds__(256) void tmod_kernel(const float* __restrict__ temb,
                                                   const float* __restrict__ Wt,
                                                   const float* __restrict__ bt,
                                                   float* __restrict__ tp) {
  int n   = blockIdx.x * 256 + threadIdx.x;      // 0 .. B*2D-1
  int b   = n / (2 * DD);
  int col = n % (2 * DD);
  float acc = bt[col];
  for (int k = 0; k < DD; ++k)
    acc += gelu_f(temb[b * DD + k]) * Wt[(size_t)k * (2 * DD) + col];
  tp[n] = acc;
}

// ---------- LayerNorm (+ optional AdaLN modulation), bf16 out ----------
template <int MOD>
__global__ __launch_bounds__(256) void ln_kernel(const float* __restrict__ x,
                                                 const float* __restrict__ g,
                                                 const float* __restrict__ beta,
                                                 const float* __restrict__ tp,
                                                 unsigned short* __restrict__ out) {
  __shared__ float red[256];
  const int row = blockIdx.x;                    // 0 .. B*L-1
  const int b   = row / LL;
  const int tid = threadIdx.x;
  const float* xr = x + (size_t)row * DD;

  float vals[4];
  float sum = 0.f;
#pragma unroll
  for (int i = 0; i < 4; ++i) { vals[i] = xr[tid + 256 * i]; sum += vals[i]; }
  red[tid] = sum; __syncthreads();
  for (int s = 128; s > 0; s >>= 1) { if (tid < s) red[tid] += red[tid + s]; __syncthreads(); }
  const float mu = red[0] * (1.0f / DD);
  __syncthreads();

  float sq = 0.f;
#pragma unroll
  for (int i = 0; i < 4; ++i) { float d = vals[i] - mu; sq += d * d; }
  red[tid] = sq; __syncthreads();
  for (int s = 128; s > 0; s >>= 1) { if (tid < s) red[tid] += red[tid + s]; __syncthreads(); }
  const float rstd = rsqrtf(red[0] * (1.0f / DD) + 1e-5f);

#pragma unroll
  for (int i = 0; i < 4; ++i) {
    int idx = tid + 256 * i;
    float hv = (vals[i] - mu) * rstd * g[idx] + beta[idx];
    if (MOD) hv = hv * (1.0f + tp[b * 2 * DD + idx]) + tp[b * 2 * DD + DD + idx];
    out[(size_t)row * DD + idx] = f2bf(hv);
  }
}

// ---------- tiled WMMA GEMM: C[M,N] = A[M,K](bf16) * Bt[N,K](bf16, transposed) ----
// Block tile 128x64, BK=32; 256 threads = 8 waves; wave w owns rows [16w,16w+16).
// Async double-buffered staging; all fragment loads are contiguous b128 runs.
template <int HAS_BIAS, int ACT_GELU, int HAS_RES, int OUT_BF16>
__global__ __launch_bounds__(256) void gemm_kernel(const unsigned short* __restrict__ A,
                                                   const unsigned short* __restrict__ Bt,
                                                   const float* __restrict__ bias,
                                                   const float* __restrict__ res,
                                                   void* __restrict__ outp,
                                                   int M, int N, int K) {
  __shared__ __align__(16) unsigned short As[2][128][40];   // 32 used + pad
  __shared__ __align__(16) unsigned short BsT[2][64][40];   // [n][k], 32 used + pad
  const int tid  = threadIdx.x;
  const int lane = tid & 31;
  const int w    = tid >> 5;
  const int half = lane >> 4;
  const int l15  = lane & 15;
  const int m0   = blockIdx.y * 128;
  const int n0   = blockIdx.x * 64;

  v8f acc[4];
#pragma unroll
  for (int i = 0; i < 4; ++i) acc[i] = v8f_zero();

  const int arow = tid >> 1;            // 0..127, 2 threads/row, 16 elems each
  const int acol = (tid & 1) * 16;
  const int brow = tid >> 2;            // 0..63 (n), 4 threads/row, 8 elems each
  const int bcol = (tid & 3) * 8;       // k chunk

  auto stage = [&](int buf, int k0) {
    const unsigned short* ag = A + (size_t)(m0 + arow) * K + k0 + acol;
    async_b128(ag,     &As[buf][arow][acol]);
    async_b128(ag + 8, &As[buf][arow][acol + 8]);
    const unsigned short* bg = Bt + (size_t)(n0 + brow) * K + k0 + bcol;
    async_b128(bg, &BsT[buf][brow][bcol]);
  };

  stage(0, 0);
  for (int k0 = 0; k0 < K; k0 += 32) {
    const int  cur  = (k0 >> 5) & 1;
    const bool more = (k0 + 32 < K);
    if (more) stage(cur ^ 1, k0 + 32);
    if (more) { WAIT_ASYNCCNT(0x3); } else { WAIT_ASYNCCNT(0x0); }
    __syncthreads();

    // A fragment: rows 16w + (lane&15); one contiguous 16-elem run per half
    v16u af;
    {
      const int r  = 16 * w + l15;
      const int kb = half * 16;
#pragma unroll
      for (int v = 0; v < 8; ++v) {
        int k = kb + 2 * v;
        af[2 * v]     = As[cur][r][k];
        af[2 * v + 1] = As[cur][r][k + 1];
      }
    }
#pragma unroll
    for (int nt = 0; nt < 4; ++nt) {
      // B fragment: lane -> column n, contiguous K run of 16 (per-half)
      v16u bf;
      const unsigned short* bp = &BsT[cur][nt * 16 + l15][half * 16];
#pragma unroll
      for (int v = 0; v < 8; ++v) { bf[2 * v] = bp[2 * v]; bf[2 * v + 1] = bp[2 * v + 1]; }
      acc[nt] = wmma_bf16(af, bf, acc[nt]);
    }
    __syncthreads();
  }

  // epilogue: C layout lane<16 -> n=lane, m=v ; lane>=16 -> n=lane-16, m=v+8
#pragma unroll
  for (int nt = 0; nt < 4; ++nt) {
#pragma unroll
    for (int v = 0; v < 8; ++v) {
      const int gm = m0 + 16 * w + v + half * 8;
      const int gn = n0 + nt * 16 + l15;
      float val = acc[nt][v];
      if (HAS_BIAS) val += bias[gn];
      if (ACT_GELU) val = gelu_f(val);
      if (HAS_RES)  val += res[(size_t)gm * N + gn];
      if (OUT_BF16) ((unsigned short*)outp)[(size_t)gm * N + gn] = f2bf(val);
      else          ((float*)outp)[(size_t)gm * N + gn] = val;
    }
  }
}

// ---------- flash attention ----------
// Block = 128 threads = 4 waves sharing (b,h); wave w owns query tile qb0+w.
// K/V 32x64 tiles async-staged per block with double buffering.
__global__ __launch_bounds__(128) void attn_kernel(const unsigned short* __restrict__ qkv,
                                                   const float* __restrict__ torus,
                                                   const float* __restrict__ tscale,
                                                   const unsigned char* __restrict__ mask,
                                                   unsigned short* __restrict__ attn_out) {
  __shared__ __align__(16) unsigned short Kt[2][32][72];
  __shared__ __align__(16) unsigned short Vt[2][32][72];
  __shared__ __align__(16) unsigned short Pbuf[4][16][40];  // per-wave 16x32 P tile
  const int tid  = threadIdx.x;
  const int lane = tid & 31;
  const int w    = tid >> 5;
  const int half = lane >> 4;
  const int l15  = lane & 15;
  const int widx = blockIdx.x * 4 + w;
  const int qb   = widx & 127;                   // L/16 = 128
  const int h    = (widx >> 7) & 15;
  const int b    = widx >> 11;
  const float ts = tscale[h];
  const int qrow0 = qb * 16;

  auto stageKV = [&](int buf, int kb0) {
#pragma unroll
    for (int c = 0; c < 2; ++c) {
      const int chunk = tid + c * 128;           // 0..255 : 8 chunks per row
      const int r   = chunk >> 3;                // key row 0..31
      const int col = (chunk & 7) * 8;           // dim col (8 ushorts = 16B)
      const size_t rowbase = (size_t)(b * LL + kb0 + r) * (3 * DD) + h * HD + col;
      async_b128(qkv + rowbase + DD,     &Kt[buf][r][col]);
      async_b128(qkv + rowbase + 2 * DD, &Vt[buf][r][col]);
    }
  };

  // Q fragments (16x64 -> two 16x32 A-fragments), loaded once from global
  v16u qf[2];
  {
    const unsigned short* qp =
        qkv + (size_t)(b * LL + qrow0 + l15) * (3 * DD) + h * HD;
    const int kb = half * 16;
#pragma unroll
    for (int f = 0; f < 2; ++f) {
#pragma unroll
      for (int v = 0; v < 8; ++v) {
        int k = f * 32 + kb + 2 * v;
        qf[f][2 * v]     = qp[k];
        qf[f][2 * v + 1] = qp[k + 1];
      }
    }
  }

  v8f accO[4];
#pragma unroll
  for (int i = 0; i < 4; ++i) accO[i] = v8f_zero();
  float mrow[8], lrow[8];
#pragma unroll
  for (int v = 0; v < 8; ++v) { mrow[v] = -1e30f; lrow[v] = 0.f; }

  stageKV(0, 0);
  for (int kb0 = 0; kb0 < LL; kb0 += 32) {
    const int  cur  = (kb0 >> 5) & 1;
    const bool more = (kb0 + 32 < LL);
    if (more) stageKV(cur ^ 1, kb0 + 32);        // overlap next tile with compute
    if (more) { WAIT_ASYNCCNT(0x4); } else { WAIT_ASYNCCNT(0x0); }
    __syncthreads();

    // ---- S = Q * K^T for 32 keys (two 16x16 tiles) ----
    v8f s[2];
#pragma unroll
    for (int t = 0; t < 2; ++t) {
      v8f c = v8f_zero();
#pragma unroll
      for (int s2 = 0; s2 < 2; ++s2) {
        v16u kf;  // B-fragment of K^T: lane -> key, contiguous head-dim run
        const unsigned short* kp = &Kt[cur][t * 16 + l15][s2 * 32 + half * 16];
#pragma unroll
        for (int v = 0; v < 8; ++v) { kf[2 * v] = kp[2 * v]; kf[2 * v + 1] = kp[2 * v + 1]; }
        c = wmma_bf16(qf[s2], kf, c);
      }
      s[t] = c;
    }

    // ---- bias + mask + online softmax (rows live per 16-lane half) ----
    float bm[8];
#pragma unroll
    for (int v = 0; v < 8; ++v) {
      const int m  = v + half * 8;
      const int qi = qrow0 + m;
      float s0 = s[0][v] * 0.125f - torus[(size_t)qi * LL + kb0 + l15] * ts;
      float s1 = s[1][v] * 0.125f - torus[(size_t)qi * LL + kb0 + 16 + l15] * ts;
      if (!mask[b * LL + kb0 + l15])      s0 = -1e30f;
      if (!mask[b * LL + kb0 + 16 + l15]) s1 = -1e30f;
      s[0][v] = s0; s[1][v] = s1;
      bm[v] = fmaxf(s0, s1);
    }
#pragma unroll
    for (int off = 1; off < 16; off <<= 1)
#pragma unroll
      for (int v = 0; v < 8; ++v) bm[v] = fmaxf(bm[v], __shfl_xor(bm[v], off, 32));

    float alpha[8], rs[8];
#pragma unroll
    for (int v = 0; v < 8; ++v) {
      float mnew = fmaxf(mrow[v], bm[v]);
      alpha[v] = __expf(mrow[v] - mnew);
      mrow[v] = mnew;
      float p0 = (s[0][v] <= -1e29f) ? 0.f : __expf(s[0][v] - mnew);
      float p1 = (s[1][v] <= -1e29f) ? 0.f : __expf(s[1][v] - mnew);
      s[0][v] = p0; s[1][v] = p1;
      rs[v] = p0 + p1;
    }
#pragma unroll
    for (int off = 1; off < 16; off <<= 1)
#pragma unroll
      for (int v = 0; v < 8; ++v) rs[v] += __shfl_xor(rs[v], off, 32);
#pragma unroll
    for (int v = 0; v < 8; ++v) lrow[v] = lrow[v] * alpha[v] + rs[v];
#pragma unroll
    for (int nt = 0; nt < 4; ++nt)
#pragma unroll
      for (int v = 0; v < 8; ++v) accO[nt][v] *= alpha[v];

    // ---- re-layout P (C-frag) -> A-frag via LDS ----
    __syncthreads();
#pragma unroll
    for (int t = 0; t < 2; ++t)
#pragma unroll
      for (int v = 0; v < 8; ++v)
        Pbuf[w][v + half * 8][t * 16 + l15] = f2bf(s[t][v]);
    __syncthreads();

    v16u pf;
    {
      const int kb = half * 16;
#pragma unroll
      for (int v = 0; v < 8; ++v) {
        int k = kb + 2 * v;
        pf[2 * v]     = Pbuf[w][l15][k];
        pf[2 * v + 1] = Pbuf[w][l15][k + 1];
      }
    }

    // ---- O += P * V  (V fragments from LDS) ----
#pragma unroll
    for (int nt = 0; nt < 4; ++nt) {
      v16u vf;  // B-fragment of V: lane -> dim col, VGPR pair -> key rows
      const int dim = nt * 16 + l15;
#pragma unroll
      for (int v = 0; v < 8; ++v) {
        vf[2 * v]     = Vt[cur][half * 16 + 2 * v][dim];
        vf[2 * v + 1] = Vt[cur][half * 16 + 2 * v + 1][dim];
      }
      accO[nt] = wmma_bf16(pf, vf, accO[nt]);
    }
    __syncthreads();   // buffer `cur` reads done before it is re-staged next iter
  }

  // ---- normalize + store (nan_to_num: empty rows -> 0) ----
#pragma unroll
  for (int nt = 0; nt < 4; ++nt)
#pragma unroll
    for (int v = 0; v < 8; ++v) {
      const int m = v + half * 8;
      const float inv = (lrow[v] > 0.f) ? (1.0f / lrow[v]) : 0.f;
      attn_out[(size_t)(b * LL + qrow0 + m) * DD + h * HD + nt * 16 + l15] =
          f2bf(accO[nt][v] * inv);
    }
}

// ------------------------------------------------------------------
extern "C" void kernel_launch(void* const* d_in, const int* in_sizes, int n_in,
                              void* d_out, int out_size, void* d_ws, size_t ws_size,
                              hipStream_t stream) {
  const float* x      = (const float*)d_in[0];
  const float* torus  = (const float*)d_in[1];
  const float* temb   = (const float*)d_in[2];
  const unsigned char* mask = (const unsigned char*)d_in[3];
  const float* ln1_g  = (const float*)d_in[4];
  const float* ln1_b  = (const float*)d_in[5];
  const float* Wqkv   = (const float*)d_in[6];
  const float* Wout   = (const float*)d_in[7];
  const float* tsc    = (const float*)d_in[8];
  const float* ln2_g  = (const float*)d_in[9];
  const float* ln2_b  = (const float*)d_in[10];
  const float* W1     = (const float*)d_in[11];
  const float* b1     = (const float*)d_in[12];
  const float* W2     = (const float*)d_in[13];
  const float* b2     = (const float*)d_in[14];
  const float* Wt     = (const float*)d_in[15];
  const float* bt     = (const float*)d_in[16];
  float* out = (float*)d_out;

  char* ws = (char*)d_ws;
  size_t off = 0;
  auto alloc = [&](size_t bytes) -> void* {
    void* p = ws + off;
    off += (bytes + 255) & ~(size_t)255;
    return p;
  };
  float*          tp     = (float*)alloc((size_t)BB * 2 * DD * 4);
  unsigned short* hbuf   = (unsigned short*)alloc((size_t)BLROWS * DD * 2);
  unsigned short* qkvb   = (unsigned short*)alloc((size_t)BLROWS * 3 * DD * 2);
  unsigned short* attnb  = (unsigned short*)alloc((size_t)BLROWS * DD * 2);
  unsigned short* h2buf  = (unsigned short*)alloc((size_t)BLROWS * DD * 2);
  unsigned short* midb   = (unsigned short*)alloc((size_t)BLROWS * FF * 2);
  unsigned short* wqkv_t = (unsigned short*)alloc((size_t)DD * 3 * DD * 2);  // [3D][D]
  unsigned short* wout_t = (unsigned short*)alloc((size_t)DD * DD * 2);      // [D][D]
  unsigned short* w1_t   = (unsigned short*)alloc((size_t)DD * FF * 2);      // [F][D]
  unsigned short* w2_t   = (unsigned short*)alloc((size_t)FF * DD * 2);      // [D][F]

  // weights -> bf16, transposed to [N][K] for contiguous B-fragment runs
  cvt_t_kernel<<<dim3((3 * DD) / 32, DD / 32), 256, 0, stream>>>(Wqkv, wqkv_t, DD, 3 * DD);
  cvt_t_kernel<<<dim3(DD / 32, DD / 32),       256, 0, stream>>>(Wout, wout_t, DD, DD);
  cvt_t_kernel<<<dim3(FF / 32, DD / 32),       256, 0, stream>>>(W1,   w1_t,   DD, FF);
  cvt_t_kernel<<<dim3(DD / 32, FF / 32),       256, 0, stream>>>(W2,   w2_t,   FF, DD);

  // AdaLN time modulation
  tmod_kernel<<<(BB * 2 * DD) / 256, 256, 0, stream>>>(temb, Wt, bt, tp);

  // h = LN1(x)*(1+scale)+shift   (bf16)
  ln_kernel<1><<<BLROWS, 256, 0, stream>>>(x, ln1_g, ln1_b, tp, hbuf);

  // qkv = h @ Wqkv               (bf16)
  gemm_kernel<0, 0, 0, 1><<<dim3((3 * DD) / 64, BLROWS / 128), 256, 0, stream>>>(
      hbuf, wqkv_t, nullptr, nullptr, qkvb, BLROWS, 3 * DD, DD);

  // flash attention -> attnb     (bf16)
  attn_kernel<<<(BB * HH * (LL / 16)) / 4, 128, 0, stream>>>(qkvb, torus, tsc, mask, attnb);

  // out = x + attn @ Wout        (f32, into d_out)
  gemm_kernel<0, 0, 1, 0><<<dim3(DD / 64, BLROWS / 128), 256, 0, stream>>>(
      attnb, wout_t, nullptr, x, out, BLROWS, DD, DD);

  // h2 = LN2(out)                (bf16)
  ln_kernel<0><<<BLROWS, 256, 0, stream>>>(out, ln2_g, ln2_b, nullptr, h2buf);

  // mid = gelu(h2 @ W1 + b1)     (bf16)
  gemm_kernel<1, 1, 0, 1><<<dim3(FF / 64, BLROWS / 128), 256, 0, stream>>>(
      h2buf, w1_t, b1, nullptr, midb, BLROWS, FF, DD);

  // out = out + mid @ W2 + b2    (f32)
  gemm_kernel<1, 0, 1, 0><<<dim3(DD / 64, BLROWS / 128), 256, 0, stream>>>(
      midb, w2_t, b2, out, out, BLROWS, DD, FF);
}
